// Lstm_unit_17600775979868
// MI455X (gfx1250) — compile-verified, matrix operand order checked by reading, not run on hardware
//
#include <hip/hip_runtime.h>

// Problem constants (match reference)
#define B_ROWS   262144
#define D_IN     64
#define D_H      128

#define NWAVES   16
#define BLOCK_T  (NWAVES * 32)          // 512 threads, 16 waves (wave32)
#define ROWS_PER_BLOCK (NWAVES * 16)    // 256 batch rows per block iteration
#define GRID_BLOCKS 512                 // 512*256 = 131072 rows/iter -> 2 iters

// Padded LDS row strides (halfs) to spread half-wave lane addresses across banks
#define WK_PAD   72                     // W rows: 64 -> 72  (144B: 16 distinct banks)
#define VK_PAD   136                    // V rows: 128 -> 136 (272B: 16 distinct banks)

typedef __bf16 v16bf __attribute__((ext_vector_type(16)));
typedef float  v8f   __attribute__((ext_vector_type(8)));

union FragU {
  v16bf v;
  __bf16 b[16];
  uint4 q[2];
};

__device__ __forceinline__ float frcp(float x) { return __builtin_amdgcn_rcpf(x); }

__device__ __forceinline__ v8f wmma_bf16(v16bf a, v16bf b, v8f c) {
  // (neg_a, A, neg_b, B, c_mod, C, reuse_a, reuse_b)
  return __builtin_amdgcn_wmma_f32_16x16x32_bf16(false, a, false, b, (short)0, c,
                                                 false, false);
}

// A fragment (16x32, bf16) from global f32 row-major [rows][ld].
// lanes 0..15: row m, K = k0..k0+15 ; lanes 16..31: row m, K = k0+16..k0+31
__device__ __forceinline__ v16bf loadA_global(const float* __restrict__ src, int ld,
                                              size_t row0, int k0, int lane) {
  int m  = lane & 15;
  int kh = (lane & 16);                  // 0 or 16
  const float* p = src + (row0 + (size_t)m) * (size_t)ld + k0 + kh;
  FragU f;
#pragma unroll
  for (int i = 0; i < 4; ++i) {
    float4 w = *(const float4*)(p + 4 * i);
    f.b[4*i+0] = (__bf16)w.x; f.b[4*i+1] = (__bf16)w.y;
    f.b[4*i+2] = (__bf16)w.z; f.b[4*i+3] = (__bf16)w.w;
  }
  return f.v;
}

// B fragment (32x16, bf16) from LDS bf16 matrix stored [N][stride] row-major.
// B[k][n] = W[n][k]; lane n holds K = k0..k0+15 (lanes 0-15) / +16 (lanes 16-31)
__device__ __forceinline__ v16bf loadB_lds(const __bf16* w, int stride,
                                           int n0, int k0, int lane) {
  const __bf16* p = w + (size_t)(n0 + (lane & 15)) * (size_t)stride + k0 + (lane & 16);
  FragU f;
  f.q[0] = *(const uint4*)(p);
  f.q[1] = *(const uint4*)(p + 8);
  return f.v;
}

__global__ __launch_bounds__(BLOCK_T)
void lstm_wmma_kernel(const float* __restrict__ x,
                      const float* __restrict__ ph,
                      const float* __restrict__ pm,
                      const float* __restrict__ Wi_w, const float* __restrict__ Wi_b,
                      const float* __restrict__ Vi_w, const float* __restrict__ Vi_b,
                      const float* __restrict__ Wf_w, const float* __restrict__ Wf_b,
                      const float* __restrict__ Vf_w, const float* __restrict__ Vf_b,
                      const float* __restrict__ Wo_w, const float* __restrict__ Wo_b,
                      const float* __restrict__ Vo_w, const float* __restrict__ Vo_b,
                      float* __restrict__ outCt, float* __restrict__ outH)
{
  // bf16 weight copies, [N][padded K] row-major
  __shared__ __align__(16) __bf16 sWi[D_H * WK_PAD];
  __shared__ __align__(16) __bf16 sWf[D_H * WK_PAD];
  __shared__ __align__(16) __bf16 sWo[D_H * WK_PAD];
  __shared__ __align__(16) __bf16 sVi[D_H * VK_PAD];
  __shared__ __align__(16) __bf16 sVf[D_H * VK_PAD];
  __shared__ __align__(16) __bf16 sVo[D_H * VK_PAD];
  __shared__ float sCbI[D_H], sCbF[D_H], sCbO[D_H];
  // per-wave staging: F values ([m][n], same-lane write/read) and
  // Fh = F*prev_hidden ([m][k] for the D->A transpose)
  __shared__ __align__(16) __bf16 sStgF [NWAVES][16 * VK_PAD];
  __shared__ __align__(16) __bf16 sStgFh[NWAVES][16 * VK_PAD];

  const int tid  = threadIdx.x;
  const int lane = tid & 31;
  const int wv   = tid >> 5;
  const int hsel = (lane >> 4);          // 0: rows r, 1: rows r+8 of the D tile

  for (int i = tid; i < D_H * D_IN; i += BLOCK_T) {
    int r = i >> 6, c = i & 63;
    int o = r * WK_PAD + c;
    sWi[o] = (__bf16)Wi_w[i]; sWf[o] = (__bf16)Wf_w[i]; sWo[o] = (__bf16)Wo_w[i];
  }
  for (int i = tid; i < D_H * D_H; i += BLOCK_T) {
    int r = i >> 7, c = i & 127;
    int o = r * VK_PAD + c;
    sVi[o] = (__bf16)Vi_w[i]; sVf[o] = (__bf16)Vf_w[i]; sVo[o] = (__bf16)Vo_w[i];
  }
  for (int i = tid; i < D_H; i += BLOCK_T) {
    sCbI[i] = Wi_b[i] + Vi_b[i];
    sCbF[i] = Wf_b[i] + Vf_b[i];
    sCbO[i] = Wo_b[i] + Vo_b[i];
  }
  __syncthreads();

  const int iters = B_ROWS / (gridDim.x * ROWS_PER_BLOCK);
  __bf16* stF  = &sStgF [wv][0];
  __bf16* stFh = &sStgFh[wv][0];

  for (int it = 0; it < iters; ++it) {
    size_t rowBase = ((size_t)it * gridDim.x + blockIdx.x) * ROWS_PER_BLOCK
                   + (size_t)wv * 16;

    // ---- A fragments: x tile (K=64) and prev_hidden tile (K=128) ----
    v16bf Ax[2], Ah[4];
#pragma unroll
    for (int kk = 0; kk < 2; ++kk) Ax[kk] = loadA_global(x,  D_IN, rowBase, 32 * kk, lane);
#pragma unroll
    for (int kk = 0; kk < 4; ++kk) Ah[kk] = loadA_global(ph, D_H,  rowBase, 32 * kk, lane);

    __syncthreads();   // previous iteration's staging reads are done

    // ---- F = sigmoid(x@Wf^T + h@Vf^T + bf); fused staging of F and F*h ----
#pragma unroll
    for (int t = 0; t < 8; ++t) {
      // batch all 6 B fragments so their ds_loads can overlap
      v16bf bF[6];
      bF[0] = loadB_lds(sWf, WK_PAD, 16 * t, 0,  lane);
      bF[1] = loadB_lds(sWf, WK_PAD, 16 * t, 32, lane);
#pragma unroll
      for (int kk = 0; kk < 4; ++kk)
        bF[2 + kk] = loadB_lds(sVf, VK_PAD, 16 * t, 32 * kk, lane);

      v8f a = {};
      a = wmma_bf16(Ax[0], bF[0], a);
      a = wmma_bf16(Ax[1], bF[1], a);
#pragma unroll
      for (int kk = 0; kk < 4; ++kk) a = wmma_bf16(Ah[kk], bF[2 + kk], a);

      int n = (lane & 15) + 16 * t;
      float cb = sCbF[n];
      // batched exp pass then rcp pass: independent trans ops fill TRANS stalls
      float e[8];
#pragma unroll
      for (int r = 0; r < 8; ++r) e[r] = __expf(-(a[r] + cb));
#pragma unroll
      for (int r = 0; r < 8; ++r) {
        float Fr = frcp(1.0f + e[r]);
        int m = r + 8 * hsel;
        float phv = ph[(rowBase + m) * D_H + n];
        stF [m * VK_PAD + n] = (__bf16)Fr;          // same-lane readback later
        stFh[m * VK_PAD + n] = (__bf16)(Fr * phv);  // cross-lane transpose source
      }
    }
    __syncthreads();   // cross-lane visibility of the transposed tile

    v16bf Afh[4];
#pragma unroll
    for (int kk = 0; kk < 4; ++kk) Afh[kk] = loadB_lds(stFh, VK_PAD, 0, 32 * kk, lane);

    // ---- per N-tile: wi_x, I, C (sharing Vi fragments), O, combine, store ----
#pragma unroll
    for (int t = 0; t < 8; ++t) {
      // Phase 1: wi_x + I + C (Wi and Vi fragments, Vi shared by I and C)
      v16bf bIC[6];
      bIC[0] = loadB_lds(sWi, WK_PAD, 16 * t, 0,  lane);
      bIC[1] = loadB_lds(sWi, WK_PAD, 16 * t, 32, lane);
#pragma unroll
      for (int kk = 0; kk < 4; ++kk)
        bIC[2 + kk] = loadB_lds(sVi, VK_PAD, 16 * t, 32 * kk, lane);

      v8f aW = {};
      aW = wmma_bf16(Ax[0], bIC[0], aW);
      aW = wmma_bf16(Ax[1], bIC[1], aW);
      v8f aI = aW, aC = aW;
#pragma unroll
      for (int kk = 0; kk < 4; ++kk) {
        aI = wmma_bf16(Ah[kk],  bIC[2 + kk], aI);
        aC = wmma_bf16(Afh[kk], bIC[2 + kk], aC);
      }

      // Phase 2: O (Wo and Vo fragments)
      v16bf bO[6];
      bO[0] = loadB_lds(sWo, WK_PAD, 16 * t, 0,  lane);
      bO[1] = loadB_lds(sWo, WK_PAD, 16 * t, 32, lane);
#pragma unroll
      for (int kk = 0; kk < 4; ++kk)
        bO[2 + kk] = loadB_lds(sVo, VK_PAD, 16 * t, 32 * kk, lane);

      v8f aO = {};
      aO = wmma_bf16(Ax[0], bO[0], aO);
      aO = wmma_bf16(Ax[1], bO[1], aO);
#pragma unroll
      for (int kk = 0; kk < 4; ++kk) aO = wmma_bf16(Ah[kk], bO[2 + kk], aO);

      int n = (lane & 15) + 16 * t;
      float cbi = sCbI[n], cbo = sCbO[n];

      // ---- batched activation epilogue (exp passes, rcp passes, combines) ----
      float tI[8], tC[8], tO[8];
#pragma unroll
      for (int r = 0; r < 8; ++r) tI[r] = __expf(-(aI[r] + cbi));
#pragma unroll
      for (int r = 0; r < 8; ++r) tC[r] = __expf(-(aC[r] + cbi));
#pragma unroll
      for (int r = 0; r < 8; ++r) tO[r] = __expf(-(aO[r] + cbo));
#pragma unroll
      for (int r = 0; r < 8; ++r) tI[r] = frcp(1.0f + tI[r]);
#pragma unroll
      for (int r = 0; r < 8; ++r) tC[r] = frcp(1.0f + tC[r]);
#pragma unroll
      for (int r = 0; r < 8; ++r) tO[r] = frcp(1.0f + tO[r]);

      float ctv[8], eT[8];
#pragma unroll
      for (int r = 0; r < 8; ++r) {
        int m = r + 8 * hsel;
        size_t row = rowBase + m;
        float pmv = pm[row * D_H + n];
        float Fr  = (float)stF[m * VK_PAD + n];     // same-lane readback
        ctv[r] = Fr * pmv + tI[r] * tC[r];
      }
#pragma unroll
      for (int r = 0; r < 8; ++r) eT[r] = __expf(2.0f * ctv[r]);
#pragma unroll
      for (int r = 0; r < 8; ++r) {
        size_t row = rowBase + r + 8 * hsel;
        float th  = 1.0f - 2.0f * frcp(1.0f + eT[r]);   // tanh(ct)
        outCt[row * D_H + n] = ctv[r];
        outH [row * D_H + n] = tO[r] * th;
      }
    }
  }
}

extern "C" void kernel_launch(void* const* d_in, const int* in_sizes, int n_in,
                              void* d_out, int out_size, void* d_ws, size_t ws_size,
                              hipStream_t stream) {
  const float* x    = (const float*)d_in[0];
  const float* ph   = (const float*)d_in[1];
  const float* pm   = (const float*)d_in[2];
  const float* Wi_w = (const float*)d_in[3];
  const float* Wi_b = (const float*)d_in[4];
  const float* Vi_w = (const float*)d_in[5];
  const float* Vi_b = (const float*)d_in[6];
  const float* Wf_w = (const float*)d_in[7];
  const float* Wf_b = (const float*)d_in[8];
  const float* Vf_w = (const float*)d_in[9];
  const float* Vf_b = (const float*)d_in[10];
  const float* Wo_w = (const float*)d_in[11];
  const float* Wo_b = (const float*)d_in[12];
  const float* Vo_w = (const float*)d_in[13];
  const float* Vo_b = (const float*)d_in[14];

  float* outCt = (float*)d_out;
  float* outH  = outCt + (size_t)B_ROWS * D_H;

  hipLaunchKernelGGL(lstm_wmma_kernel, dim3(GRID_BLOCKS), dim3(BLOCK_T), 0, stream,
                     x, ph, pm, Wi_w, Wi_b, Vi_w, Vi_b, Wf_w, Wf_b, Vf_w, Vf_b,
                     Wo_w, Wo_b, Vo_w, Vo_b, outCt, outH);
}
